// AttentionWithRotaryPosEmb_33217277067421
// MI455X (gfx1250) — matrix operand order repaired and checked by reading, not compiled, not run on hardware
//
#include <hip/hip_runtime.h>
#include <hip/hip_bf16.h>
#include <math.h>

// ---------------------------------------------------------------------------
// AttentionWithRotaryPosEmb for MI455X (gfx1250), f16 WMMA pipeline.
// b=8, C=256, H=8, D=64, s=1024.  All GEMMs use v_wmma_f32_16x16x32_f16.
// Weight panels are staged into LDS with the Tensor Data Mover (TENSORcnt).
// ---------------------------------------------------------------------------

typedef __attribute__((ext_vector_type(16))) _Float16 v16h;
typedef __attribute__((ext_vector_type(8)))  _Float16 v8h;
typedef __attribute__((ext_vector_type(8)))  float    v8f;
typedef __attribute__((ext_vector_type(4)))  unsigned u32x4;
typedef __attribute__((ext_vector_type(8)))  int      i32x8;
typedef __attribute__((ext_vector_type(4)))  int      i32x4;

#define B_      8
#define C_      256
#define HEADS_  8
#define DH_     64
#define S_      1024
#define HID_    512        // HEADS_*DH_
#define QKV_M   1536       // 3*HID_
#define SCALE_  10.0f
#define LDP_    1032       // padded LDS row stride (floats), 1032 % 64 = 8 -> no bank conflict

// ---------------- WMMA helpers -------------------------------------------

__device__ __forceinline__ v8f wmma_f16(v16h a, v16h b, v8f c) {
  // D = A(16x32 f16) x B(32x16 f16) + C(16x16 f32)
  return __builtin_amdgcn_wmma_f32_16x16x32_f16(
      /*neg_a=*/false, a, /*neg_b=*/false, b,
      /*c_mod=*/(short)0, c, /*reuse_a=*/false, /*reuse_b=*/false);
}

// A fragment from a row-major (M x K) f16 matrix with leading dim ld.
// lane l<16 : row l,     K = k0+{0..7, 16..23}
// lane l>=16: row l-16,  K = k0+{8..15,24..31}
// Works for both global and LDS source pointers (inlined -> addrspace inferred).
__device__ __forceinline__ v16h frag_a_f16(const _Float16* base, int row0, int ld, int k0) {
  const int l  = threadIdx.x & 31;
  const int r  = l & 15;
  const int ko = (l & 16) ? 8 : 0;
  const _Float16* p = base + (size_t)(row0 + r) * ld + k0 + ko;
  v8h lo = *(const v8h*)(p);        // 16B aligned by construction
  v8h hi = *(const v8h*)(p + 16);
  v16h out;
#pragma unroll
  for (int i = 0; i < 8; ++i) { out[i] = lo[i]; out[i + 8] = hi[i]; }
  return out;
}

// B fragment for B(32x16) taken from B^T stored row-major (N x K, leading dim ld).
// lane l: col = l&15, K = k0+{0..7,16..23} (l<16) or k0+{8..15,24..31} (l>=16).
__device__ __forceinline__ v16h frag_b_f16(const _Float16* baseT, int n0, int ld, int k0) {
  const int l  = threadIdx.x & 31;
  const int n  = l & 15;
  const int ko = (l & 16) ? 8 : 0;
  const _Float16* p = baseT + (size_t)(n0 + n) * ld + k0 + ko;
  v8h lo = *(const v8h*)(p);
  v8h hi = *(const v8h*)(p + 16);
  v16h out;
#pragma unroll
  for (int i = 0; i < 8; ++i) { out[i] = lo[i]; out[i + 8] = hi[i]; }
  return out;
}

// A fragment built from an f32 LDS matrix (softmax probs), scaled per-row, cvt to f16.
__device__ __forceinline__ v16h frag_a_lds_scaled(const float* S, int row0, int ld,
                                                  int k0, const float* invsum) {
  const int l  = threadIdx.x & 31;
  const int r  = l & 15;
  const int ko = (l & 16) ? 8 : 0;
  const float* p = S + (row0 + r) * ld + k0 + ko;
  const float iv = invsum[row0 + r];
  v16h out;
#pragma unroll
  for (int i = 0; i < 8; ++i) {
    out[i]     = (_Float16)(p[i]      * iv);
    out[i + 8] = (_Float16)(p[i + 16] * iv);
  }
  return out;
}

// ---------------- Tensor Data Mover: stage contiguous f16 panel to LDS ---
// Panel is (rows x cols) f16, row stride == cols (contiguous block).
// D# layout per CDNA5 ISA (2D tile, groups 2/3 zero).

#if __has_builtin(__builtin_amdgcn_tensor_load_to_lds)
__device__ __forceinline__ void tdm_load_panel_f16(_Float16* lds, const _Float16* g,
                                                   unsigned rows, unsigned cols) {
  const unsigned long long ga = (unsigned long long)g;
  const unsigned lds_off = (unsigned)(unsigned long long)lds;   // low 32b = LDS offset
  u32x4 g0;
  g0[0] = 1u;                                                // count=1, user descriptor
  g0[1] = lds_off;                                           // lds_addr
  g0[2] = (unsigned)ga;                                      // global_addr[31:0]
  g0[3] = (unsigned)((ga >> 32) & 0x01FFFFFFull) | (2u << 30); // addr[56:32] | type=2
  i32x8 g1;
  g1[0] = (int)(1u << 16);                                   // data_size=1 -> 2B elements
  g1[1] = (int)((cols & 0xFFFFu) << 16);                     // tensor_dim0[15:0]
  g1[2] = (int)((cols >> 16) | ((rows & 0xFFFFu) << 16));    // dim0[31:16] | dim1[15:0]
  g1[3] = (int)((rows >> 16) | ((cols & 0xFFFFu) << 16));    // dim1[31:16] | tile_dim0
  g1[4] = (int)(rows & 0xFFFFu);                             // tile_dim1 (tile_dim2=0)
  g1[5] = (int)cols;                                         // tensor_dim0_stride[31:0]
  g1[6] = 0;
  g1[7] = 0;
  i32x4 gz = {0, 0, 0, 0};
#if defined(__clang_major__) && (__clang_major__ >= 23)
  i32x8 gz8 = {0, 0, 0, 0, 0, 0, 0, 0};
  __builtin_amdgcn_tensor_load_to_lds(g0, g1, gz, gz, gz8, 0);
#else
  __builtin_amdgcn_tensor_load_to_lds(g0, g1, gz, gz, 0);
#endif
}
#endif

__device__ __forceinline__ void stage_panel(_Float16* lds, const _Float16* g,
                                            unsigned rows, unsigned cols) {
#if __has_builtin(__builtin_amdgcn_tensor_load_to_lds)
  if (threadIdx.x < 32) {                 // one wave issues the DMA (EXEC ignored by TDM)
    tdm_load_panel_f16(lds, g, rows, cols);
    __builtin_amdgcn_s_wait_tensorcnt(0); // only issuing wave's TENSORcnt counts
  }
#else
  const unsigned n = rows * cols;
  for (unsigned i = threadIdx.x; i < n; i += 256u) lds[i] = g[i];
#endif
  __syncthreads();                        // publish LDS panel to all 8 waves
}

// ---------------- 1) prep: convert/transposes ----------------------------

__global__ __launch_bounds__(256) void k_prep(const float* __restrict__ x,
                                              const float* __restrict__ w_qkv,
                                              const float* __restrict__ w_out,
                                              _Float16* __restrict__ xT,
                                              _Float16* __restrict__ wqkv_h,
                                              _Float16* __restrict__ wout_h) {
  const int idx = blockIdx.x * 256 + threadIdx.x;
  // x (b, C, s) f32  ->  xT (b, s, C) f16   (x read exactly once: non-temporal)
  if (idx < B_ * S_ * C_) {
    const int b = idx >> 18;                 // / (S_*C_)
    const int rem = idx & (S_ * C_ - 1);
    const int s = rem >> 8;                  // / C_
    const int c = rem & (C_ - 1);
    const float v = __builtin_nontemporal_load(&x[((size_t)b << 18) + (size_t)c * S_ + s]);
    xT[idx] = (_Float16)v;
  }
  if (idx < QKV_M * C_) wqkv_h[idx] = (_Float16)w_qkv[idx];
  if (idx < C_ * HID_)  wout_h[idx] = (_Float16)w_out[idx];
}

// ---------------- 2) QKV GEMM: (1536x256) x (256x1024) per batch ---------
// A panel (64x256 f16, 32KB) staged to LDS by TDM; B streamed from global.

__global__ __launch_bounds__(256) void k_qkv_gemm(const _Float16* __restrict__ wqkv_h,
                                                  const _Float16* __restrict__ xT,
                                                  float* __restrict__ qkv_raw) {
  __shared__ _Float16 Apanel[64 * C_];
  const int mblk = blockIdx.x;            // 0..23  (64 rows each)
  const int nblk = blockIdx.y;            // 0..7   (128 cols each)
  const int b    = blockIdx.z;            // 0..7
  const int w    = threadIdx.x >> 5;      // wave id 0..7
  const int m0l  = (w >> 1) * 16;         // row inside LDS panel
  const int nb   = nblk * 128 + (w & 1) * 64;
  const _Float16* Bb = xT + (size_t)b * S_ * C_;

  stage_panel(Apanel, wqkv_h + (size_t)(mblk * 64) * C_, 64, C_);

  v8f c[4] = {};
  for (int k0 = 0; k0 < C_; k0 += 32) {
    __builtin_prefetch(Bb + (size_t)nb * C_ + k0 + 32, 0, 3);
    v16h a = frag_a_f16(Apanel, m0l, C_, k0);           // ds_load path
#pragma unroll
    for (int nt = 0; nt < 4; ++nt) {
      v16h bf = frag_b_f16(Bb, nb + nt * 16, C_, k0);   // global_load path
      c[nt] = wmma_f16(a, bf, c[nt]);
    }
  }
  float* outp = qkv_raw + (size_t)b * QKV_M * S_;
  const int m0 = mblk * 64 + m0l;
  const int l = threadIdx.x & 31;
  const int col = l & 15;
  const int rb = (l & 16) ? 8 : 0;
#pragma unroll
  for (int nt = 0; nt < 4; ++nt)
#pragma unroll
    for (int r = 0; r < 8; ++r)
      outp[(size_t)(m0 + rb + r) * S_ + nb + nt * 16 + col] = c[nt][r];
}

// ---------------- 3) RoPE + l2norm(s) + layout conversions ---------------
// qkv_raw rows per (b): [0,512)=q  [512,1024)=k  [1024,1536)=v
// Outputs: qh,kh (b,h, s,64) f16;  vh (b,h, 64,s) f16

__device__ __forceinline__ float rope_val(const float* row, const float* prow,
                                          int d, int i, float invfreq) {
  float v = row[i];
  if (d < 32) {
    float sn, cs;
    __sincosf((float)i * invfreq, &sn, &cs);
    const float p = prow[i];
    v = (d < 16) ? (v * cs - p * sn) : (v * cs + p * sn);
  }
  return v;
}

__device__ void rope_norm_one(const float* __restrict__ src,   // (64, 1024) d-major
                              _Float16* __restrict__ dst,      // (1024, 64) s-major
                              float* red, float* invn) {
  const int t = threadIdx.x;
  const int d = t & 63;
  const int i0 = (t >> 6) * 256;
  const int j = (d < 16) ? d : (d - 16);
  const float invfreq = __expf(-(float)j * 0.575646273248511f);  // ln(10000)/16
  const float* row  = src + (size_t)d * S_;
  const int pd = (d < 16) ? d + 16 : ((d < 32) ? d - 16 : d);
  const float* prow = src + (size_t)pd * S_;

  float ss = 0.f;
  for (int i = i0; i < i0 + 256; ++i) {
    const float v = rope_val(row, prow, d, i, invfreq);
    ss += v * v;
  }
  red[t] = ss;
  __syncthreads();
  if (t < 64) {
    const float tot = red[t] + red[t + 64] + red[t + 128] + red[t + 192];
    invn[t] = 1.0f / fmaxf(sqrtf(tot), 1e-12f);
  }
  __syncthreads();
  const float iv = invn[d];
  for (int i = i0; i < i0 + 256; ++i) {
    const float v = rope_val(row, prow, d, i, invfreq);
    dst[(size_t)i * DH_ + d] = (_Float16)(v * iv);
  }
  __syncthreads();
}

__global__ __launch_bounds__(256) void k_rope_norm(const float* __restrict__ qkv_raw,
                                                   _Float16* __restrict__ qh,
                                                   _Float16* __restrict__ kh,
                                                   _Float16* __restrict__ vh) {
  __shared__ float red[256];
  __shared__ float invn[64];
  const int bh = blockIdx.x;           // 0..63
  const int b = bh >> 3, h = bh & 7;
  const float* base = qkv_raw + (size_t)b * QKV_M * S_;
  const float* qsrc = base + (size_t)(h * DH_) * S_;
  const float* ksrc = base + (size_t)(HID_ + h * DH_) * S_;
  const float* vsrc = base + (size_t)(2 * HID_ + h * DH_) * S_;

  // v: fp32 (64,1024) -> f16 (64,1024), coalesced
  _Float16* vdst = vh + (size_t)bh * DH_ * S_;
  for (int idx = threadIdx.x; idx < DH_ * S_; idx += 256)
    vdst[idx] = (_Float16)vsrc[idx];

  rope_norm_one(qsrc, qh + (size_t)bh * S_ * DH_, red, invn);
  rope_norm_one(ksrc, kh + (size_t)bh * S_ * DH_, red, invn);
}

// ---------------- 4) attention: S=Q^T K (LDS) -> softmax -> O=P V^T ------
// One block per (b,h, 64-query rows). 256 threads = 8 waves. 264KB dynamic LDS
// (CDNA5 WGP has 320KB): whole 64x1024 score block lives in LDS, so softmax
// needs no online/flash bookkeeping.

__global__ __launch_bounds__(256) void k_attn(const _Float16* __restrict__ qh,
                                              const _Float16* __restrict__ kh,
                                              const _Float16* __restrict__ vh,
                                              _Float16* __restrict__ attn_outT) {
  extern __shared__ float S[];          // [64][LDP_] f32 scores, then probs
  __shared__ float invsum[64];
  const int mblk = blockIdx.x;          // 0..15
  const int bh   = blockIdx.y;          // 0..63
  const int b = bh >> 3, h = bh & 7;
  const _Float16* qB = qh + (size_t)bh * S_ * DH_ + (size_t)mblk * 64 * DH_;
  const _Float16* kB = kh + (size_t)bh * S_ * DH_;
  const _Float16* vB = vh + (size_t)bh * DH_ * S_;
  const int w = threadIdx.x >> 5;
  const int l = threadIdx.x & 31;
  const int col = l & 15;
  const int rb = (l & 16) ? 8 : 0;

  // ---- phase 1: scores for this wave's 128-column strip -----------------
#pragma unroll
  for (int mt = 0; mt < 4; ++mt) {
    const int m0 = mt * 16;
    v16h a0 = frag_a_f16(qB, m0, DH_, 0);
    v16h a1 = frag_a_f16(qB, m0, DH_, 32);
#pragma unroll
    for (int nt = 0; nt < 8; ++nt) {
      const int n0 = w * 128 + nt * 16;
      v16h b0 = frag_b_f16(kB, n0, DH_, 0);
      v16h b1 = frag_b_f16(kB, n0, DH_, 32);
      v8f c = {};
      c = wmma_f16(a0, b0, c);
      c = wmma_f16(a1, b1, c);
#pragma unroll
      for (int r = 0; r < 8; ++r)
        S[(m0 + rb + r) * LDP_ + n0 + col] = c[r] * SCALE_;
    }
  }
  __syncthreads();

  // ---- phase 2: softmax over 1024 keys, 4 lanes per row (wave32 shfl) ---
  {
    const int t = threadIdx.x;
    const int r = t >> 2, qq = t & 3;
    float* row = S + r * LDP_ + qq * 256;
    float m = -1e30f;
    for (int jj = 0; jj < 256; ++jj) m = fmaxf(m, row[jj]);
    m = fmaxf(m, __shfl_xor(m, 1, 32));
    m = fmaxf(m, __shfl_xor(m, 2, 32));
    float sum = 0.f;
    for (int jj = 0; jj < 256; ++jj) {
      const float e = __expf(row[jj] - m);
      row[jj] = e;
      sum += e;
    }
    sum += __shfl_xor(sum, 1, 32);
    sum += __shfl_xor(sum, 2, 32);
    if (qq == 0) invsum[r] = 1.0f / sum;
  }
  __syncthreads();

  // ---- phase 3: O = P (64x1024) x V^T (1024x64), 2 output tiles / wave --
  const int m0 = (w >> 1) * 16;
  const int n0 = (w & 1) * 32;
  v8f c0 = {}, c1 = {};
  for (int k0 = 0; k0 < S_; k0 += 32) {
    v16h a  = frag_a_lds_scaled(S, m0, LDP_, k0, invsum);
    v16h b0 = frag_b_f16(vB, n0, S_, k0);
    v16h b1 = frag_b_f16(vB, n0 + 16, S_, k0);
    c0 = wmma_f16(a, b0, c0);
    c1 = wmma_f16(a, b1, c1);
  }
  // attn_outT (b, s, 512) f16; channel = h*64 + d
  _Float16* outp = attn_outT + ((size_t)b * S_ + (size_t)mblk * 64) * HID_ + h * DH_;
#pragma unroll
  for (int r = 0; r < 8; ++r) {
    const int i = m0 + rb + r;
    outp[(size_t)i * HID_ + n0 + col]      = (_Float16)c0[r];
    outp[(size_t)i * HID_ + n0 + 16 + col] = (_Float16)c1[r];
  }
}

// ---------------- 5) output projection: (256x512) x (512x1024) + bias ----
// A panel (64x512 f16, 64KB) staged to LDS by TDM.

__global__ __launch_bounds__(256) void k_out_proj(const _Float16* __restrict__ wout_h,
                                                  const _Float16* __restrict__ attn_outT,
                                                  const float* __restrict__ b_out,
                                                  float* __restrict__ out) {
  __shared__ _Float16 Apanel[64 * HID_];
  const int mblk = blockIdx.x;            // 0..3
  const int nblk = blockIdx.y;            // 0..7
  const int b    = blockIdx.z;            // 0..7
  const int w    = threadIdx.x >> 5;
  const int m0l  = (w >> 1) * 16;
  const int nb   = nblk * 128 + (w & 1) * 64;
  const _Float16* Bb = attn_outT + (size_t)b * S_ * HID_;

  stage_panel(Apanel, wout_h + (size_t)(mblk * 64) * HID_, 64, HID_);

  v8f c[4] = {};
  for (int k0 = 0; k0 < HID_; k0 += 32) {
    __builtin_prefetch(Bb + (size_t)nb * HID_ + k0 + 32, 0, 3);
    v16h a = frag_a_f16(Apanel, m0l, HID_, k0);
#pragma unroll
    for (int nt = 0; nt < 4; ++nt) {
      v16h bf = frag_b_f16(Bb, nb + nt * 16, HID_, k0);
      c[nt] = wmma_f16(a, bf, c[nt]);
    }
  }
  float* outp = out + (size_t)b * C_ * S_;
  const int m0 = mblk * 64 + m0l;
  const int l = threadIdx.x & 31;
  const int col = l & 15;
  const int rb = (l & 16) ? 8 : 0;
#pragma unroll
  for (int nt = 0; nt < 4; ++nt)
#pragma unroll
    for (int r = 0; r < 8; ++r) {
      const int o = m0 + rb + r;
      outp[(size_t)o * S_ + nb + nt * 16 + col] = c[nt][r] + b_out[o];
    }
}

// ---------------- launcher -----------------------------------------------

extern "C" void kernel_launch(void* const* d_in, const int* in_sizes, int n_in,
                              void* d_out, int out_size, void* d_ws, size_t ws_size,
                              hipStream_t stream) {
  (void)in_sizes; (void)n_in; (void)out_size; (void)ws_size;
  const float* x     = (const float*)d_in[0];   // (8,256,32,32)
  const float* w_qkv = (const float*)d_in[1];   // (1536,256)
  const float* w_out = (const float*)d_in[2];   // (256,512)
  const float* b_out = (const float*)d_in[3];   // (256,)
  float* out = (float*)d_out;                   // (8,256,32,32)

  // workspace carve-out (~85 MB total)
  char* ws = (char*)d_ws;
  _Float16* wqkv_h   = (_Float16*)ws; ws += (size_t)QKV_M * C_ * 2;        //  768 KB
  _Float16* wout_h   = (_Float16*)ws; ws += (size_t)C_ * HID_ * 2;         //  256 KB
  _Float16* xT       = (_Float16*)ws; ws += (size_t)B_ * S_ * C_ * 2;      //    4 MB
  float*    qkv_raw  = (float*)ws;    ws += (size_t)B_ * QKV_M * S_ * 4;   //   48 MB
  _Float16* qh       = (_Float16*)ws; ws += (size_t)B_ * HEADS_ * S_ * DH_ * 2; // 8 MB
  _Float16* kh       = (_Float16*)ws; ws += (size_t)B_ * HEADS_ * S_ * DH_ * 2; // 8 MB
  _Float16* vh       = (_Float16*)ws; ws += (size_t)B_ * HEADS_ * DH_ * S_ * 2; // 8 MB
  _Float16* attn_outT= (_Float16*)ws; ws += (size_t)B_ * S_ * HID_ * 2;         // 8 MB

  // 1) convert weights / transpose x to f16
  k_prep<<<(B_ * S_ * C_) / 256, 256, 0, stream>>>(x, w_qkv, w_out, xT, wqkv_h, wout_h);

  // 2) QKV projection (TDM-staged weight panel)
  k_qkv_gemm<<<dim3(QKV_M / 64, S_ / 128, B_), 256, 0, stream>>>(wqkv_h, xT, qkv_raw);

  // 3) RoPE + l2 norm + v convert
  k_rope_norm<<<B_ * HEADS_, 256, 0, stream>>>(qkv_raw, qh, kh, vh);

  // 4) attention (264 KB dynamic LDS per workgroup; CDNA5 WGP has 320 KB)
  k_attn<<<dim3(S_ / 64, B_ * HEADS_), 256, (size_t)64 * LDP_ * sizeof(float), stream>>>(
      qh, kh, vh, attn_outT);

  // 5) output projection + bias (TDM-staged weight panel)
  k_out_proj<<<dim3(C_ / 64, S_ / 128, B_), 256, 0, stream>>>(wout_h, attn_outT, b_out, out);
}
